// NobsRelation_75857712382390
// MI455X (gfx1250) — matrix operand order repaired and checked by reading, not compiled, not run on hardware
//
#include <hip/hip_runtime.h>
#include <hip/hip_bf16.h>

// Problem constants (from reference): N=1024, D_FEAT=HID=128, K_REL=8, C_REL=64
#define NN   1024
#define DF   128
#define KREL 8
#define CREL 64
#define BI   32   // block tile rows (i)
#define BJ   64   // block tile cols (j)

typedef __attribute__((ext_vector_type(16))) _Float16 v16h;
typedef __attribute__((ext_vector_type(8)))  _Float16 v8h;
typedef __attribute__((ext_vector_type(8)))  float    v8f;

// ---------------------------------------------------------------------------
// Kernel 1: left = x @ W1[:128] + b1, right = x @ W1[128:], stored as f16.
// ---------------------------------------------------------------------------
__global__ void nobs_prep_lr(const float* __restrict__ x,
                             const float* __restrict__ W1,
                             const float* __restrict__ b1,
                             _Float16* __restrict__ left_h,
                             _Float16* __restrict__ right_h) {
    __shared__ float sx[DF];
    const int i = blockIdx.x;
    const int c = threadIdx.x;
    sx[c] = x[i * DF + c];
    __syncthreads();
    float aL = b1[c];   // fold b1 into the i-side contribution
    float aR = 0.0f;
    #pragma unroll 8
    for (int d = 0; d < DF; ++d) {
        const float xv = sx[d];
        aL += xv * W1[d * DF + c];
        aR += xv * W1[(DF + d) * DF + c];
    }
    left_h[i * DF + c]  = (_Float16)aL;
    right_h[i * DF + c] = (_Float16)aR;
}

// ---------------------------------------------------------------------------
// Kernel 2: E f32 -> f16 cast; W2 [128,8] -> padded transpose W2T [16][128].
// ---------------------------------------------------------------------------
__global__ void nobs_prep_ew(const float* __restrict__ E,
                             const float* __restrict__ W2,
                             _Float16* __restrict__ E_h,
                             _Float16* __restrict__ W2T) {
    const int idx = blockIdx.x * 256 + threadIdx.x;
    if (idx < KREL * NN * CREL) E_h[idx] = (_Float16)E[idx];
    if (idx < 16 * DF) {
        const int nn = idx / DF;     // padded output column (k), 0..15
        const int cc = idx % DF;     // hidden index
        W2T[idx] = (nn < KREL) ? (_Float16)W2[cc * KREL + nn] : (_Float16)0.0f;
    }
}

// ---------------------------------------------------------------------------
// Async global->LDS tile fill (CDNA5 GLOBAL_LOAD_ASYNC_TO_LDS_B128, GVS mode,
// tracked by ASYNCcnt). lds_off is a byte offset into the (dynamic) LDS
// allocation; gsrc is the 64-bit SGPR base; per-lane 32-bit byte offset.
// ---------------------------------------------------------------------------
__device__ __forceinline__ void async_fill16(unsigned lds_off_base,
                                             const void* gsrc,
                                             int n_chunks16, int tid) {
    for (int i = tid; i < n_chunks16; i += 256) {
        const unsigned loff = lds_off_base + (unsigned)i * 16u;
        const unsigned goff = (unsigned)i * 16u;
        asm volatile("global_load_async_to_lds_b128 %0, %1, %2 offset:0"
                     :: "v"(loff), "v"(goff), "s"(gsrc)
                     : "memory");
    }
}

// ---------------------------------------------------------------------------
// Fused main kernel. 256 threads = 8 waves tiling 2x4 over a 32x64 block;
// each wave computes one 16x16 subtile of A.
// ---------------------------------------------------------------------------
__global__ void __launch_bounds__(256)
nobs_main(const _Float16* __restrict__ left_h,
          const _Float16* __restrict__ right_h,
          const _Float16* __restrict__ E_h,
          const _Float16* __restrict__ W2T,
          const float*    __restrict__ b2,
          const float*    __restrict__ gumbel,
          float*          __restrict__ out) {
    extern __shared__ char smem[];
    // LDS carve-up (all offsets 256B aligned)
    _Float16* sL  = (_Float16*)(smem);                  // [32][128]   8192 B
    _Float16* sR  = (_Float16*)(smem + 8192);           // [64][128]  16384 B
    _Float16* sEi = (_Float16*)(smem + 24576);          // [8][32][64] 32768 B
    _Float16* sEj = (_Float16*)(smem + 57344);          // [8][64][64] 65536 B
    _Float16* sW2 = (_Float16*)(smem + 122880);         // [16][128]   4096 B
    float*    sD  = (float*)   (smem + 126976);         // 8 waves * [8][16][16]
    float*    sLg = (float*)   (smem + 192512);         // 8 waves * [2][16][16]

    const int tid   = threadIdx.x;
    const int wave  = tid >> 5;
    const int lane  = tid & 31;
    const int nlan  = lane & 15;   // row/col within 16 (WMMA M/N per lane)
    const int hg    = lane >> 4;   // K half-group selector per ISA layout
    const int i_blk = blockIdx.y * BI;
    const int j_blk = blockIdx.x * BJ;
    const int wi    = wave >> 2;   // 0..1 -> i sub-tile
    const int wj    = wave & 3;    // 0..3 -> j sub-tile

    // ---- async memory->LDS tile fill (ASYNCcnt path) ----
    async_fill16(0u,      left_h  + (size_t)i_blk * DF, 512, tid);
    async_fill16(8192u,   right_h + (size_t)j_blk * DF, 1024, tid);
    async_fill16(122880u, W2T, 256, tid);
    for (int k = 0; k < KREL; ++k) {
        async_fill16(24576u + (unsigned)k * 4096u,
                     E_h + ((size_t)k * NN + i_blk) * CREL, 256, tid);
        async_fill16(57344u + (unsigned)k * 8192u,
                     E_h + ((size_t)k * NN + j_blk) * CREL, 512, tid);
    }
    asm volatile("s_wait_asynccnt 0" ::: "memory");
    __syncthreads();

    float* sDw  = sD  + wave * (KREL * 256);
    float* sLgw = sLg + wave * (2 * 256);

    // ---- Phase 1: D[k] = E[k,i-tile] @ E[k,j-tile]^T via WMMA f16 ----
    #pragma unroll
    for (int k = 0; k < KREL; ++k) {
        v8f acc = {};
        const _Float16* Ei = sEi + (k * BI + wi * 16 + nlan) * CREL; // A row = i
        const _Float16* Ej = sEj + (k * BJ + wj * 16 + nlan) * CREL; // B col = j
        #pragma unroll
        for (int cc = 0; cc < 2; ++cc) {           // K = 64 -> two 16x16x32
            v16h a, b;
            ((v8h*)&a)[0] = *(const v8h*)(Ei + cc * 32 + hg * 8);
            ((v8h*)&a)[1] = *(const v8h*)(Ei + cc * 32 + 16 + hg * 8);
            ((v8h*)&b)[0] = *(const v8h*)(Ej + cc * 32 + hg * 8);
            ((v8h*)&b)[1] = *(const v8h*)(Ej + cc * 32 + 16 + hg * 8);
            acc = __builtin_amdgcn_wmma_f32_16x16x32_f16(
                false, a, false, b, (short)0, acc, false, false);
        }
        #pragma unroll
        for (int v = 0; v < 8; ++v)
            sDw[(k * 16 + (v + hg * 8)) * 16 + nlan] = acc[v];  // [k][m=i][n=j]
    }

    // ---- Phase 2: per i-row logits GEMM + gumbel softmax + dot with D ----
    const float b2v = (nlan < KREL) ? b2[nlan] : 0.0f;
    const _Float16* Rrow = sR + (wj * 16 + nlan) * DF;   // h A-frag row = j
    const _Float16* Wrow = sW2 + nlan * DF;              // W2^T B-frag col = k

    // Loop-invariant fragments: right-half of h operand, and the W2 B operand.
    v16h rfrag[4], wfrag[4];
    #pragma unroll
    for (int cb = 0; cb < 4; ++cb) {
        const int c0 = cb * 32 + hg * 8;
        ((v8h*)&rfrag[cb])[0] = *(const v8h*)(Rrow + c0);
        ((v8h*)&rfrag[cb])[1] = *(const v8h*)(Rrow + c0 + 16);
        const int cw = cb * 32 + hg * 8;
        ((v8h*)&wfrag[cb])[0] = *(const v8h*)(Wrow + cw);
        ((v8h*)&wfrag[cb])[1] = *(const v8h*)(Wrow + cw + 16);
    }

    for (int ii0 = 0; ii0 < 16; ii0 += 2) {
        #pragma unroll
        for (int t = 0; t < 2; ++t) {
            const int ii = ii0 + t;
            const _Float16* Lrow = sL + (wi * 16 + ii) * DF;
            v8f lacc = {};
            #pragma unroll
            for (int cb = 0; cb < 4; ++cb) {        // K = 128 -> four 16x16x32
                const int c0 = cb * 32 + hg * 8;
                const v8h l0 = *(const v8h*)(Lrow + c0);      // lane-broadcast
                const v8h l1 = *(const v8h*)(Lrow + c0 + 16);
                const v8h r0 = ((const v8h*)&rfrag[cb])[0];
                const v8h r1 = ((const v8h*)&rfrag[cb])[1];
                v16h a;
                #pragma unroll
                for (int e = 0; e < 8; ++e) {       // h = relu(left+right), f16
                    _Float16 s0 = (_Float16)(l0[e] + r0[e]);
                    _Float16 s1 = (_Float16)(l1[e] + r1[e]);
                    a[e]     = s0 > (_Float16)0.0f ? s0 : (_Float16)0.0f;
                    a[8 + e] = s1 > (_Float16)0.0f ? s1 : (_Float16)0.0f;
                }
                lacc = __builtin_amdgcn_wmma_f32_16x16x32_f16(
                    false, a, false, wfrag[cb], (short)0, lacc, false, false);
            }
            #pragma unroll
            for (int v = 0; v < 8; ++v) {           // relu(logits + b2), stage
                float lv = lacc[v] + b2v;
                lv = lv > 0.0f ? lv : 0.0f;
                sLgw[((t * 16) + v + hg * 8) * 16 + nlan] = lv;  // [t][m=j][n=k]
            }
        }

        // combine: lane -> pair (ii = ii0+hg, jj = nlan)
        const int gi = i_blk + wi * 16 + ii0 + hg;
        const int gj = j_blk + wj * 16 + nlan;
        const float* gmb = gumbel + ((size_t)gi * NN + (size_t)gj) * KREL;
        if (ii0 < 14) __builtin_prefetch(gmb + (size_t)2 * NN * KREL, 0, 0);
        const float4 g0 = *(const float4*)gmb;
        const float4 g1 = *(const float4*)(gmb + 4);
        const float* lgp = sLgw + (hg * 16 + nlan) * 16;  // logits[k], k=0..7
        float z[8];
        z[0] = lgp[0] + g0.x; z[1] = lgp[1] + g0.y;
        z[2] = lgp[2] + g0.z; z[3] = lgp[3] + g0.w;
        z[4] = lgp[4] + g1.x; z[5] = lgp[5] + g1.y;
        z[6] = lgp[6] + g1.z; z[7] = lgp[7] + g1.w;
        float mx = z[0];
        #pragma unroll
        for (int k = 1; k < KREL; ++k) mx = z[k] > mx ? z[k] : mx;
        float ssum = 0.0f, ez[8];
        #pragma unroll
        for (int k = 0; k < KREL; ++k) { ez[k] = __expf(z[k] - mx); ssum += ez[k]; }
        const float inv = 1.0f / ssum;
        float accA = 0.0f;
        #pragma unroll
        for (int k = 0; k < KREL; ++k)
            accA += sDw[(k * 16 + ii0 + hg) * 16 + nlan] * ez[k];
        out[(size_t)gi * NN + (size_t)gj] = accA * inv;
    }
}

// ---------------------------------------------------------------------------
// Host-side launch. Inputs: x, W1, b1, W2, b2, E, gumbel_noise (f32).
// ---------------------------------------------------------------------------
extern "C" void kernel_launch(void* const* d_in, const int* in_sizes, int n_in,
                              void* d_out, int out_size, void* d_ws, size_t ws_size,
                              hipStream_t stream) {
    const float* x   = (const float*)d_in[0];
    const float* W1  = (const float*)d_in[1];
    const float* b1  = (const float*)d_in[2];
    const float* W2  = (const float*)d_in[3];
    const float* b2  = (const float*)d_in[4];
    const float* E   = (const float*)d_in[5];
    const float* gum = (const float*)d_in[6];
    float* out = (float*)d_out;

    char* ws = (char*)d_ws;
    _Float16* left_h  = (_Float16*)(ws);                       // 262144 B
    _Float16* right_h = (_Float16*)(ws + 262144);              // 262144 B
    _Float16* E_h     = (_Float16*)(ws + 524288);              // 1048576 B
    _Float16* W2T     = (_Float16*)(ws + 524288 + 1048576);    // 4096 B

    nobs_prep_lr<<<NN, DF, 0, stream>>>(x, W1, b1, left_h, right_h);
    nobs_prep_ew<<<(KREL * NN * CREL) / 256, 256, 0, stream>>>(E, W2, E_h, W2T);

    dim3 grid(NN / BJ, NN / BI);   // (16, 32)
    dim3 blk(256);
    const size_t lds_bytes = 208896;  // 204 KB carve (<= 320 KB WGP budget)
    nobs_main<<<grid, blk, lds_bytes, stream>>>(left_h, right_h, E_h, W2T,
                                                b2, gum, out);
}
